// Distance_41197326303940
// MI455X (gfx1250) — compile-verified, matrix operand order checked by reading, not run on hardware
//
#include <hip/hip_runtime.h>
#include <hip/hip_bf16.h>

// ---------------------------------------------------------------------------
// Center-distance loss:
//   loss = mean_{b, c != label_b} clip(sqrt(|x_b|^2 + |c_c|^2 - 2 x_b.c_c), 1e-8, 1e8)
// B=16384, C=1000, D=256.  Dominant cost: 16384x1000x256 GEMM (~8.4 GFLOP).
// Strategy: bf16 WMMA GEMM with hi/lo split for fp32-level accuracy
// (x ~= hi + lo; dot ~= hi.hi + hi.lo + lo.hi), fused sqrt/mask epilogue,
// hierarchical reduction to a single f64 accumulator.
// ---------------------------------------------------------------------------

typedef __attribute__((ext_vector_type(16))) __bf16 v16bf;
typedef __attribute__((ext_vector_type(8)))  __bf16 v8bf;
typedef __attribute__((ext_vector_type(8)))  float  v8f;

#define BATCH 16384
#define NCLS  1000
#define ZDIM  256

#define TILE_M 128   // 8 waves x 16 rows
#define TILE_N 64    // 4 accumulators x 16 cols per wave
#define TILE_K 32    // one bf16 WMMA K-step
#define LDA    40    // padded LDS row stride (elements); 80B keeps 16B alignment

union FragU { v16bf v; v8bf h[2]; };

__device__ __forceinline__ void split_bf16(float f, __bf16& hi, __bf16& lo) {
    hi = (__bf16)f;
    lo = (__bf16)(f - (float)hi);
}

// ---------------------------------------------------------------------------
__global__ void init_accum_kernel(double* acc) {
    if (threadIdx.x == 0 && blockIdx.x == 0) *acc = 0.0;
}

// One wave per row: sum of squares over ZDIM=256 floats.
__global__ void row_sq_kernel(const float* __restrict__ src,
                              float* __restrict__ dst, int nrows) {
    int wave = threadIdx.x >> 5;
    int lane = threadIdx.x & 31;
    int row  = blockIdx.x * 8 + wave;
    if (row >= nrows) return;
    const float4* p = (const float4*)(src + (long)row * ZDIM);
    float s = 0.f;
#pragma unroll
    for (int i = 0; i < 2; ++i) {            // 256 / 32 lanes = 8 floats/lane
        float4 v = p[lane + i * 32];
        s += v.x * v.x + v.y * v.y + v.z * v.z + v.w * v.w;
    }
#pragma unroll
    for (int off = 16; off; off >>= 1) s += __shfl_down(s, off, 32);
    if (lane == 0) dst[row] = s;
}

// ---------------------------------------------------------------------------
// Main fused GEMM + distance + masked-sum kernel.
// grid = (NCLS/TILE_N rounded up, BATCH/TILE_M), block = 256 threads (8 waves).
__global__ void __launch_bounds__(256)
center_loss_kernel(const float* __restrict__ x,
                   const float* __restrict__ centers,
                   const int*   __restrict__ labels,
                   const float* __restrict__ xsq,
                   const float* __restrict__ csq,
                   double* __restrict__ acc_out) {
    __shared__ __align__(16) __bf16 a_hi[TILE_M * LDA];
    __shared__ __align__(16) __bf16 a_lo[TILE_M * LDA];
    __shared__ __align__(16) __bf16 b_hi[TILE_N * LDA];
    __shared__ __align__(16) __bf16 b_lo[TILE_N * LDA];
    __shared__ float wave_sum[8];

    const int tid  = threadIdx.x;
    const int wave = tid >> 5;
    const int lane = tid & 31;
    const int m0   = blockIdx.y * TILE_M;
    const int n0   = blockIdx.x * TILE_N;

    v8f acc0 = {}, acc1 = {}, acc2 = {}, acc3 = {};

    // A/B fragment addressing (16-bit A 16x32 layout, ISA 7.12.2):
    // lanes 0-15: row = lane, K = {0..7} (VGPR0-3) and {16..23} (VGPR4-7)
    // lanes 16-31: row = lane-15? no -> row = lane-16, K = {8..15} and {24..31}
    const int frow  = lane & 15;
    const int kbase = (lane < 16) ? 0 : 8;

    for (int k0 = 0; k0 < ZDIM; k0 += TILE_K) {
        __syncthreads();   // previous iteration's fragment reads done

        // ---- stage A tile: 128 rows x 32 cols, 1024 float4 / 256 threads ----
#pragma unroll
        for (int i = 0; i < 4; ++i) {
            int idx = tid + i * 256;
            int r   = idx >> 3;
            int c4  = (idx & 7) * 4;
            const float* gp = x + (long)(m0 + r) * ZDIM + k0 + c4;
            if (k0 + TILE_K < ZDIM) __builtin_prefetch(gp + TILE_K, 0, 0);
            float4 v = *(const float4*)gp;
            __bf16 h, l;
            split_bf16(v.x, h, l); a_hi[r*LDA + c4 + 0] = h; a_lo[r*LDA + c4 + 0] = l;
            split_bf16(v.y, h, l); a_hi[r*LDA + c4 + 1] = h; a_lo[r*LDA + c4 + 1] = l;
            split_bf16(v.z, h, l); a_hi[r*LDA + c4 + 2] = h; a_lo[r*LDA + c4 + 2] = l;
            split_bf16(v.w, h, l); a_hi[r*LDA + c4 + 3] = h; a_lo[r*LDA + c4 + 3] = l;
        }
        // ---- stage B tile: 64 centers x 32 cols, 512 float4 / 256 threads ----
#pragma unroll
        for (int i = 0; i < 2; ++i) {
            int idx = tid + i * 256;
            int r   = idx >> 3;
            int c4  = (idx & 7) * 4;
            int cr  = n0 + r;
            float4 v = make_float4(0.f, 0.f, 0.f, 0.f);
            if (cr < NCLS) v = *(const float4*)(centers + (long)cr * ZDIM + k0 + c4);
            __bf16 h, l;
            split_bf16(v.x, h, l); b_hi[r*LDA + c4 + 0] = h; b_lo[r*LDA + c4 + 0] = l;
            split_bf16(v.y, h, l); b_hi[r*LDA + c4 + 1] = h; b_lo[r*LDA + c4 + 1] = l;
            split_bf16(v.z, h, l); b_hi[r*LDA + c4 + 2] = h; b_lo[r*LDA + c4 + 2] = l;
            split_bf16(v.w, h, l); b_hi[r*LDA + c4 + 3] = h; b_lo[r*LDA + c4 + 3] = l;
        }
        __syncthreads();

        // ---- A fragments for this wave's 16-row strip ----
        const int arow = wave * 16 + frow;
        FragU Ah, Al;
        Ah.h[0] = *(const v8bf*)&a_hi[arow * LDA + kbase];
        Ah.h[1] = *(const v8bf*)&a_hi[arow * LDA + kbase + 16];
        Al.h[0] = *(const v8bf*)&a_lo[arow * LDA + kbase];
        Al.h[1] = *(const v8bf*)&a_lo[arow * LDA + kbase + 16];

        // ---- 4 N-fragments, 3 chained WMMAs each (hi.hi + hi.lo + lo.hi) ----
#define DO_FRAG(J, ACC)                                                          \
        {                                                                        \
            const int brow = (J) * 16 + frow;                                    \
            FragU Bh, Bl;                                                        \
            Bh.h[0] = *(const v8bf*)&b_hi[brow * LDA + kbase];                   \
            Bh.h[1] = *(const v8bf*)&b_hi[brow * LDA + kbase + 16];              \
            Bl.h[0] = *(const v8bf*)&b_lo[brow * LDA + kbase];                   \
            Bl.h[1] = *(const v8bf*)&b_lo[brow * LDA + kbase + 16];              \
            ACC = __builtin_amdgcn_wmma_f32_16x16x32_bf16(                       \
                false, Al.v, false, Bh.v, (short)0, ACC, false, false);          \
            ACC = __builtin_amdgcn_wmma_f32_16x16x32_bf16(                       \
                false, Ah.v, false, Bl.v, (short)0, ACC, false, false);          \
            ACC = __builtin_amdgcn_wmma_f32_16x16x32_bf16(                       \
                false, Ah.v, false, Bh.v, (short)0, ACC, false, false);          \
        }
        DO_FRAG(0, acc0)
        DO_FRAG(1, acc1)
        DO_FRAG(2, acc2)
        DO_FRAG(3, acc3)
#undef DO_FRAG
    }

    // ---- fused epilogue: dist -> sqrt -> clamp -> mask -> partial sum ----
    // C/D 16x16 f32 layout: VGPR r holds row M=r (lanes 0-15) / M=r+8 (16-31),
    // col N = lane & 15.
    const int mrowbase = m0 + wave * 16 + ((lane < 16) ? 0 : 8);
    float xs[8];
    int   lb[8];
#pragma unroll
    for (int r = 0; r < 8; ++r) {
        xs[r] = xsq[mrowbase + r];
        lb[r] = labels[mrowbase + r];
    }

    float partial = 0.f;
    const v8f accs[4] = {acc0, acc1, acc2, acc3};
#pragma unroll
    for (int j = 0; j < 4; ++j) {
        const int gn = n0 + j * 16 + frow;
        if (gn < NCLS) {
            const float cs = csq[gn];
#pragma unroll
            for (int r = 0; r < 8; ++r) {
                float dist = xs[r] + cs - 2.0f * accs[j][r];
                float d = sqrtf(fmaxf(dist, 0.f));
                d = fminf(fmaxf(d, 1e-8f), 1e8f);
                if (lb[r] == gn) d = 0.f;
                partial += d;
            }
        }
    }

#pragma unroll
    for (int off = 16; off; off >>= 1) partial += __shfl_down(partial, off, 32);
    if (lane == 0) wave_sum[wave] = partial;
    __syncthreads();
    if (tid == 0) {
        float s = 0.f;
#pragma unroll
        for (int i = 0; i < 8; ++i) s += wave_sum[i];
        atomicAdd(acc_out, (double)s);   // global_atomic_add_f64
    }
}

// ---------------------------------------------------------------------------
__global__ void finalize_kernel(const double* acc, float* out) {
    if (threadIdx.x == 0 && blockIdx.x == 0)
        out[0] = (float)(*acc / ((double)BATCH * (double)(NCLS - 1)));
}

// ---------------------------------------------------------------------------
extern "C" void kernel_launch(void* const* d_in, const int* in_sizes, int n_in,
                              void* d_out, int out_size, void* d_ws, size_t ws_size,
                              hipStream_t stream) {
    const float* x       = (const float*)d_in[0];
    const float* centers = (const float*)d_in[1];
    const int*   labels  = (const int*)d_in[2];
    float*       out     = (float*)d_out;

    // workspace layout: [0..63] f64 accumulator, then xsq, then csq
    char*   ws   = (char*)d_ws;
    double* acc  = (double*)ws;
    float*  xsq  = (float*)(ws + 64);
    float*  csq  = (float*)(ws + 64 + (size_t)BATCH * sizeof(float));

    init_accum_kernel<<<1, 32, 0, stream>>>(acc);
    row_sq_kernel<<<BATCH / 8, 256, 0, stream>>>(x, xsq, BATCH);
    row_sq_kernel<<<(NCLS + 7) / 8, 256, 0, stream>>>(centers, csq, NCLS);

    dim3 grid((NCLS + TILE_N - 1) / TILE_N, BATCH / TILE_M);  // 16 x 128
    center_loss_kernel<<<grid, 256, 0, stream>>>(x, centers, labels, xsq, csq, acc);

    finalize_kernel<<<1, 32, 0, stream>>>(acc, out);
}
